// Net_12506944766173
// MI455X (gfx1250) — matrix operand order repaired and checked by reading, not compiled, not run on hardware
//
#include <hip/hip_runtime.h>

// ---------------------------------------------------------------------------
// MoE ensemble: 32 experts (4->64->64->1 tanh MLP) + gate (3->64->32 tanh MLP)
// B = 131072.  Layer-1 (93% of FLOPs) -> v_wmma_f32_16x16x32_f16.
// Layer-0 / gate-layer-0 -> v_wmma_f32_16x16x4_f32 (exact fp32, K=4).
// tanh -> hardware v_tanh_f32.  BM=32: two 16-row A tiles share each B frag.
// ---------------------------------------------------------------------------

typedef __attribute__((ext_vector_type(16))) _Float16 v16h;
typedef __attribute__((ext_vector_type(8)))  float    v8f;
typedef __attribute__((ext_vector_type(2)))  float    v2f;

#define BM 32                     // batch rows per workgroup (2 x 16-row tiles)
#define W1P_HALVES (32*2*4*32*16) // experts x kc x nt x lanes x 16 halves
#define WGP_HALVES (2*2*32*16)    // kc x nt x lanes x 16 halves

__device__ __forceinline__ float fast_tanh(float x) {
#if __has_builtin(__builtin_amdgcn_tanhf)
    return __builtin_amdgcn_tanhf(x);            // v_tanh_f32
#else
    float e = __builtin_amdgcn_exp2f(x * 2.8853900817779268f);
    return 1.0f - 2.0f * __builtin_amdgcn_rcpf(e + 1.0f);
#endif
}

// Map a (row m, K) element of a 16x32 f16 A-tile to its fragment-ordered LDS
// offset (ISA 7.12.2: lanes 0-15 M=0..15 hold K 0-7/16-23, lanes 16-31 hold
// K 8-15/24-31; 2 K values per VGPR).  Buffer layout = [kc][lane][16].
__device__ __forceinline__ int a_frag_off(int m, int K) {
    int kc  = K >> 5;
    int kk  = K & 31;
    int blk = kk >> 3;                      // 0..3
    int g   = blk & 1;                      // lane group
    int v   = ((blk >> 1) << 2) | ((kk & 7) >> 1);
    int j   = kk & 1;
    return (kc << 9) | (((g << 4) | m) << 4) | (v << 1) | j;
}

// ---------------------------------------------------------------------------
// Pre-pack: W1 [32][64][64] f32 and Wgout [64][32] f32 -> f16 in B-fragment
// order (lane l, group g=l>>4, col n = nt*16 + l%16; VGPR v holds
// K = kc*32 + (v>>2)*16 + g*8 + 2(v&3), +1).  One contiguous v16h per lane per fragment.
// ---------------------------------------------------------------------------
__global__ void pack_weights_kernel(const float* __restrict__ W1,
                                    const float* __restrict__ Wgout,
                                    _Float16* __restrict__ w1p,
                                    _Float16* __restrict__ wgp) {
    int idx = blockIdx.x * blockDim.x + threadIdx.x;
    if (idx < W1P_HALVES) {
        int i  = idx & 15;
        int l  = (idx >> 4) & 31;
        int nt = (idx >> 9) & 3;
        int kc = (idx >> 11) & 1;
        int e  = idx >> 12;
        int g = l >> 4, n = nt * 16 + (l & 15);
        int K = kc * 32 + ((i >> 3) << 4) + g * 8 + (i & 7);
        w1p[idx] = (_Float16)W1[(e * 64 + K) * 64 + n];
    } else {
        int idx2 = idx - W1P_HALVES;
        if (idx2 < WGP_HALVES) {
            int i  = idx2 & 15;
            int l  = (idx2 >> 4) & 31;
            int nt = (idx2 >> 9) & 1;
            int kc = (idx2 >> 10) & 1;
            int g = l >> 4, n = nt * 16 + (l & 15);
            int K = kc * 32 + ((i >> 3) << 4) + g * 8 + (i & 7);
            wgp[idx2] = (_Float16)Wgout[K * 32 + n];
        }
    }
}

// ---------------------------------------------------------------------------
// Main kernel: one workgroup = 32 batch rows x all 32 experts. 8 waves;
// wave w handles experts 4w..4w+3; wave 7 also computes the gate.
// ---------------------------------------------------------------------------
__global__ void __launch_bounds__(256)
moe_ensemble_kernel(const float* __restrict__ x_in,
                    const float* __restrict__ W0,  const float* __restrict__ b0,
                    const float* __restrict__ b1,
                    const float* __restrict__ Wq,  const float* __restrict__ bq,
                    const float* __restrict__ Wg0, const float* __restrict__ bg0,
                    const float* __restrict__ bgout,
                    const _Float16* __restrict__ w1p,
                    const _Float16* __restrict__ wgp,
                    float* __restrict__ experts_out,
                    float* __restrict__ gate_out,
                    float* __restrict__ ensemble_out) {
    __shared__ float x_s[BM][4];
    __shared__ __align__(32) _Float16 h0f[8][2048]; // per-wave A-frag h0, 2 tiles
    __shared__ __align__(32) _Float16 ghf[2048];    // gate hidden A-frag, 2 tiles
    __shared__ float q_s[BM][32];
    __shared__ float gate_s[BM][32];

    const int t  = threadIdx.x;
    const int w  = t >> 5;          // wave id (uniform within wave)
    const int l  = t & 31;          // lane
    const int g  = l >> 4;          // lane group
    const int lm = l & 15;
    const long brow = (long)blockIdx.x * BM;

    if (t < BM * 4) x_s[t >> 2][t & 3] = x_in[brow * 4 + t];
    __syncthreads();

    // f32 A-fragments of the two 16-row x tiles for 16x16x4 wmma:
    // lanes 0-15: K=0,1 ; lanes 16-31: K=2,3 ; row = lane%16
    v2f xa[2];
    #pragma unroll
    for (int mt = 0; mt < 2; ++mt) {
        xa[mt][0] = x_s[mt * 16 + lm][2 * g];
        xa[mt][1] = x_s[mt * 16 + lm][2 * g + 1];
    }

    // ---------------- gate (wave 7 only; wave-uniform branch) ---------------
    if (w == 7) {
        // gate layer 0: [32,3]x[3,64] as 16x16x4 wmma with zero-padded K=3
        v2f ga[2];
        #pragma unroll
        for (int mt = 0; mt < 2; ++mt) {
            ga[mt][0] = g ? x_s[mt * 16 + lm][2] : x_s[mt * 16 + lm][0];
            ga[mt][1] = g ? 0.0f                 : x_s[mt * 16 + lm][1];
        }
        #pragma unroll
        for (int nt = 0; nt < 4; ++nt) {
            int n = nt * 16 + lm;
            // load all 3 rows unconditionally (in-bounds), select branch-free
            float r0 = Wg0[0 * 64 + n];
            float r1 = Wg0[1 * 64 + n];
            float r2 = Wg0[2 * 64 + n];
            v2f gb;
            gb[0] = g ? r2   : r0;
            gb[1] = g ? 0.0f : r1;
            float bb = bg0[n];
            #pragma unroll
            for (int mt = 0; mt < 2; ++mt) {
                v8f c;
                #pragma unroll
                for (int r = 0; r < 8; ++r) c[r] = bb;
                c = __builtin_amdgcn_wmma_f32_16x16x4_f32(
                    false, ga[mt], false, gb, (short)0, c, false, false);
                #pragma unroll
                for (int r = 0; r < 8; ++r)
                    ghf[(mt << 10) + a_frag_off(r + 8 * g, n)] =
                        (_Float16)fast_tanh(c[r]);
            }
        }
        // gate output layer: [32,64]x[64,32] via 8x wmma f16
        v8f gacc[2][2];
        #pragma unroll
        for (int nt = 0; nt < 2; ++nt) {
            float bb = bgout[nt * 16 + lm];
            #pragma unroll
            for (int mt = 0; mt < 2; ++mt)
                #pragma unroll
                for (int r = 0; r < 8; ++r) gacc[mt][nt][r] = bb;
        }
        #pragma unroll
        for (int kc = 0; kc < 2; ++kc) {
            v16h a0 = *(const v16h*)&ghf[(0 << 10) | (kc << 9) | (l << 4)];
            v16h a1 = *(const v16h*)&ghf[(1 << 10) | (kc << 9) | (l << 4)];
            #pragma unroll
            for (int nt = 0; nt < 2; ++nt) {
                v16h bf = *(const v16h*)&wgp[(((kc * 2 + nt) * 32) + l) * 16];
                gacc[0][nt] = __builtin_amdgcn_wmma_f32_16x16x32_f16(
                    false, a0, false, bf, (short)0, gacc[0][nt], false, false);
                gacc[1][nt] = __builtin_amdgcn_wmma_f32_16x16x32_f16(
                    false, a1, false, bf, (short)0, gacc[1][nt], false, false);
            }
        }
        #pragma unroll
        for (int mt = 0; mt < 2; ++mt)
            #pragma unroll
            for (int nt = 0; nt < 2; ++nt) {
                int n = nt * 16 + lm;
                #pragma unroll
                for (int r = 0; r < 8; ++r) {
                    int m = mt * 16 + r + 8 * g;     // C-layout row
                    float val = fast_tanh(gacc[mt][nt][r]);
                    gate_s[m][n] = val;
                    gate_out[(brow + m) * 32 + n] = val;
                }
            }
    }

    // ---------------- experts: wave w -> experts 4w..4w+3 -------------------
    for (int ei = 0; ei < 4; ++ei) {
        const int e = w * 4 + ei;
        const float* W0e = W0 + e * 4 * 64;

        // layer 0: [32,4]x[4,64] via 8x v_wmma_f32_16x16x4_f32 (exact fp32)
        #pragma unroll
        for (int nt = 0; nt < 4; ++nt) {
            int n = nt * 16 + lm;
            v2f wb;
            wb[0] = W0e[(2 * g) * 64 + n];       // B: lanes 0-15 K=0,1
            wb[1] = W0e[(2 * g + 1) * 64 + n];   //    lanes 16-31 K=2,3
            float bb = b0[e * 64 + n];
            #pragma unroll
            for (int mt = 0; mt < 2; ++mt) {
                v8f c;
                #pragma unroll
                for (int r = 0; r < 8; ++r) c[r] = bb;
                c = __builtin_amdgcn_wmma_f32_16x16x4_f32(
                    false, xa[mt], false, wb, (short)0, c, false, false);
                #pragma unroll
                for (int r = 0; r < 8; ++r)
                    h0f[w][(mt << 10) + a_frag_off(r + 8 * g, n)] =
                        (_Float16)fast_tanh(c[r]);
            }
        }

        // layer 1: [32,64]x[64,64] via 16x v_wmma_f32_16x16x32_f16;
        // each weight B-fragment is reused by both 16-row A tiles.
        v8f acc[2][4];
        #pragma unroll
        for (int nt = 0; nt < 4; ++nt) {
            float bb = b1[e * 64 + nt * 16 + lm];   // bias per column n
            #pragma unroll
            for (int mt = 0; mt < 2; ++mt)
                #pragma unroll
                for (int r = 0; r < 8; ++r) acc[mt][nt][r] = bb;
        }
        #pragma unroll
        for (int kc = 0; kc < 2; ++kc) {
            v16h a0 = *(const v16h*)&h0f[w][(0 << 10) | (kc << 9) | (l << 4)];
            v16h a1 = *(const v16h*)&h0f[w][(1 << 10) | (kc << 9) | (l << 4)];
            #pragma unroll
            for (int nt = 0; nt < 4; ++nt) {
                v16h bf = *(const v16h*)&w1p[((((e * 2 + kc) * 4 + nt) * 32) + l) * 16];
                acc[0][nt] = __builtin_amdgcn_wmma_f32_16x16x32_f16(
                    false, a0, false, bf, (short)0, acc[0][nt], false, false);
                acc[1][nt] = __builtin_amdgcn_wmma_f32_16x16x32_f16(
                    false, a1, false, bf, (short)0, acc[1][nt], false, false);
            }
        }

        // expert head directly from C-fragments:
        // lane owns rows r+8g at column nt*16+lm ->
        // partial[r] = sum_nt tanh(h1)*Wq[n]; reduce over the 16-lane group.
        float bqe = bq[e];
        #pragma unroll
        for (int mt = 0; mt < 2; ++mt) {
            float part[8];
            #pragma unroll
            for (int r = 0; r < 8; ++r) part[r] = 0.0f;
            #pragma unroll
            for (int nt = 0; nt < 4; ++nt) {
                float wq = Wq[e * 64 + nt * 16 + lm];
                #pragma unroll
                for (int r = 0; r < 8; ++r)
                    part[r] += fast_tanh(acc[mt][nt][r]) * wq;
            }
            #pragma unroll
            for (int off = 1; off < 16; off <<= 1) {
                #pragma unroll
                for (int r = 0; r < 8; ++r)
                    part[r] += __shfl_xor(part[r], off, 32);
            }
            if (lm == 0) {
                #pragma unroll
                for (int r = 0; r < 8; ++r)
                    q_s[mt * 16 + r + 8 * g][e] = part[r] + bqe;
            }
        }
    }
    __syncthreads();

    // ---------------- gating product + outputs ------------------------------
    #pragma unroll
    for (int i = 0; i < (BM * 32) / 256; ++i) {     // 4 elements per thread
        int idx = t + 256 * i;
        int b = idx >> 5, e = idx & 31;
        float p = q_s[b][e] * gate_s[b][e];
        experts_out[(brow + b) * 32 + e] = p;
        q_s[b][e] = p;                              // reuse for ensemble reduce
    }
    __syncthreads();

    if (t < BM) {
        float s = 0.f;
        #pragma unroll
        for (int e = 0; e < 32; ++e) s += q_s[t][e];
        ensemble_out[brow + t] = s;
    }
}

// ---------------------------------------------------------------------------
extern "C" void kernel_launch(void* const* d_in, const int* in_sizes, int n_in,
                              void* d_out, int out_size, void* d_ws, size_t ws_size,
                              hipStream_t stream) {
    const float* x_in  = (const float*)d_in[0];
    const float* W0    = (const float*)d_in[1];
    const float* b0    = (const float*)d_in[2];
    const float* W1    = (const float*)d_in[3];
    const float* b1    = (const float*)d_in[4];
    const float* Wq    = (const float*)d_in[5];
    const float* bq    = (const float*)d_in[6];
    const float* Wg0   = (const float*)d_in[7];
    const float* bg0   = (const float*)d_in[8];
    const float* Wgout = (const float*)d_in[9];
    const float* bgout = (const float*)d_in[10];

    const int B = in_sizes[0] / 4;

    _Float16* w1p = (_Float16*)d_ws;
    _Float16* wgp = w1p + W1P_HALVES;

    float* out         = (float*)d_out;
    float* experts_out = out;                       // [B,32]
    float* gate_out    = out + (size_t)B * 32;      // [B,32]
    float* ensemble    = out + (size_t)B * 64;      // [B]

    // 1) pack weights to f16 WMMA B-fragment order in workspace
    int pack_threads = W1P_HALVES + WGP_HALVES;
    pack_weights_kernel<<<(pack_threads + 255) / 256, 256, 0, stream>>>(
        W1, Wgout, w1p, wgp);

    // 2) main fused kernel: one workgroup per 32 batch rows
    moe_ensemble_kernel<<<B / BM, 256, 0, stream>>>(
        x_in, W0, b0, b1, Wq, bq, Wg0, bg0, bgout,
        w1p, wgp, experts_out, gate_out, ensemble);
}
